// CConv2d_22574348107833
// MI455X (gfx1250) — matrix operand order repaired and checked by reading, not compiled, not run on hardware
//
#include <hip/hip_runtime.h>

typedef __attribute__((ext_vector_type(16))) __bf16 v16bf;
typedef __attribute__((ext_vector_type(8)))  __bf16 v8bf;
typedef __attribute__((ext_vector_type(8)))  float  v8f;
typedef unsigned int v4u __attribute__((ext_vector_type(4)));
typedef int          v8i __attribute__((ext_vector_type(8)));
typedef int          v4i __attribute__((ext_vector_type(4)));

#define BB   32
#define CIN  256
#define COUT 256
#define HH   28
#define WW   28
#define NUME 8
#define PIX  (HH*WW)        // 784
#define KPER (CIN*9)        // 2304 (GEMM K)
#define WPS  (COUT*KPER)    // 589824 dyn-weight elems per sample
#define XPS  (30*30*CIN)    // 230400 padded NHWC elems per sample

#define PIXB    528         // LDS bytes per pixel: 512 data + 16 TDM pad (bank-conflict-free)
#define SROWS   4           // padded rows staged per block
#define STILE   (SROWS*30*CIN)            // 30720 elements per TDM tile
#define SBYTES  (STILE*2 + (STILE*2/512)*16)  // 61440 + 120*16 = 63360

// -------- 1) routing gates: GAP -> 8 logits -> sigmoid --------
__global__ void gates_kernel(const float* __restrict__ in,
                             const float* __restrict__ rw,
                             const float* __restrict__ rb,
                             float* __restrict__ gates) {
    __shared__ float se[CIN];
    const int b = blockIdx.x, t = threadIdx.x;
    const float* ch = in + (size_t)(b*CIN + t)*PIX;
    float s = 0.f;
    for (int i = 0; i < PIX; i += 4) {                 // 784 % 4 == 0
        float4 v = *(const float4*)(ch + i);
        s += v.x + v.y + v.z + v.w;
    }
    se[t] = s * (1.0f/(float)PIX);
    __syncthreads();
    if (t < NUME) {
        float acc = rb[t];
        const float* w = rw + t*CIN;
        for (int c = 0; c < CIN; ++c) acc += se[c]*w[c];
        gates[b*NUME + t] = 1.f/(1.f + __expf(-acc));
    }
}

// -------- 2) input NCHW f32 -> zero-padded (30x30) NHWC bf16 --------
__global__ void nhwc_kernel(const float* __restrict__ in, __bf16* __restrict__ xp) {
    const int by = blockIdx.x;          // b*30 + y
    const int b = by / 30, y = by % 30;
    const int c = threadIdx.x;          // 256 threads = channels (coalesced writes)
    for (int x = 0; x < 30; ++x) {
        float v = 0.f;
        if (y >= 1 && y <= HH && x >= 1 && x <= WW)
            v = in[((size_t)(b*CIN + c)*HH + (y-1))*WW + (x-1)];
        xp[((size_t)(b*30 + y)*30 + x)*CIN + c] = (__bf16)v;
    }
}

// -------- 3) dynamic weights, bf16, layout [b][co][tap][ci] --------
__global__ void wsyn_kernel(const float* __restrict__ weight,
                            const float* __restrict__ gates,
                            __bf16* __restrict__ wdyn) {
    __shared__ float g[BB*NUME];        // 256 floats
    const int t = threadIdx.x;
    g[t] = gates[t];
    __syncthreads();
    const int o   = blockIdx.x*256 + t;           // < 589824
    const int co  = o / KPER;
    const int rem = o - co*KPER;
    const int tap = rem >> 8;
    const int ci  = rem & 255;
    const int r   = co*KPER + ci*9 + tap;         // row in reference (out,in,kh,kw) layout
    const float4* wp = (const float4*)(weight + (size_t)r*NUME);  // 32B aligned
    const float4 w0 = wp[0], w1 = wp[1];
    for (int b = 0; b < BB; ++b) {
        const float* gb = g + b*NUME;
        float acc = gb[0]*w0.x + gb[1]*w0.y + gb[2]*w0.z + gb[3]*w0.w
                  + gb[4]*w1.x + gb[5]*w1.y + gb[6]*w1.z + gb[7]*w1.w;
        wdyn[(size_t)b*WPS + o] = (__bf16)acc;    // coalesced per-sample
    }
}

// -------- 4) implicit-GEMM conv: TDM -> LDS staging + v_wmma_f32_16x16x32_bf16 --------
__global__ void __launch_bounds__(256) cconv_wmma_kernel(
        const __bf16* __restrict__ xp, const __bf16* __restrict__ wdyn,
        const float* __restrict__ bias, float* __restrict__ out) {
    __shared__ __align__(64) char sB[SBYTES];   // 4 padded NHWC rows, TDM-padded layout

    const int blk = blockIdx.x;
    const int b  = blk / 49;
    const int nt = blk % 49;            // 49 pixel tiles of 16 (784 = 49*16)
    const int p0 = nt*16;
    const int y0 = p0 / WW;             // first output row of this tile

    const int wid  = threadIdx.x >> 5;  // 8 waves, wave32
    const int lane = threadIdx.x & 31;
    const int l16  = lane & 15;
    const int hi   = lane >> 4;

    // ---- TDM: stage padded rows [y0, y0+4) of sample b into LDS (wave 0 issues) ----
    if (wid == 0) {
        unsigned ldsOff = (unsigned)(unsigned long long)(const void*)sB; // flat LDS addr: low 32 = LDS offset
        unsigned long long ga = (unsigned long long)(const void*)
                                (xp + (size_t)b*XPS + (size_t)y0*30*CIN);
        const unsigned td0 = (unsigned)((30 - y0) * 30 * CIN); // remaining elems; TDM zero-fills OOB tail
        v4u g0 = { 1u,                                         // count=1 (valid descriptor)
                   ldsOff,                                     // lds_addr [63:32]
                   (unsigned)ga,                               // global_addr lo
                   (unsigned)((ga >> 32) & 0x01FFFFFFu) | 0x80000000u }; // addr hi | type=2
        v8i g1 = { (int)0x07910000,                            // data_size=2B, pad_en,
                                                               // pad_interval=512B, pad_amount=16B
                   (int)((td0 & 0xFFFFu) << 16),               // tensor_dim0[15:0]
                   (int)((td0 >> 16) | 0x00010000u),           // tensor_dim0[31:16] | tensor_dim1=1
                   (int)(STILE << 16),                         // tile_dim0 = 30720
                   1,                                          // tile_dim1 = 1
                   (int)td0,                                   // tensor_dim0_stride[31:0]
                   0, 0 };
        v4i gz4 = {0, 0, 0, 0};
        v8i gz8 = {0, 0, 0, 0, 0, 0, 0, 0};
        __builtin_amdgcn_tensor_load_to_lds(g0, g1, gz4, gz4, gz8, 0);
        __builtin_amdgcn_s_wait_tensorcnt(0);
    }
    __syncthreads();

    const int p = p0 + l16;             // this lane's output pixel (B/D column)
    const int y = p / WW, x = p % WW;   // y - y0 in {0,1}

    const __bf16* wb = wdyn + (size_t)b*WPS;
    const int m0 = wid*32;              // two 16-row M tiles per wave: m0, m0+16
    // A fragment: row = lane%16; K chunks at +hi*8 and +16+hi*8 (ISA 16-bit A layout)
    const __bf16* arow0 = wb + (size_t)(m0      + l16)*KPER + hi*8;
    const __bf16* arow1 = wb + (size_t)(m0 + 16 + l16)*KPER + hi*8;

    v8f acc0 = {}, acc1 = {};
    for (int tap = 0; tap < 9; ++tap) {
        const int kh = tap/3, kw = tap - kh*3;
        // B fragment from LDS: col = lane%16 (pixel), K chunks at +hi*8 and +16+hi*8 channels
        const char* bpix = sB + (size_t)(((y - y0 + kh)*30) + (x + kw))*PIXB + hi*16;
        const __bf16* a0 = arow0 + tap*CIN;
        const __bf16* a1 = arow1 + tap*CIN;
#pragma unroll 2
        for (int c = 0; c < CIN; c += 32) {
            v8bf bl  = *(const v8bf*)(bpix + 2*c);        // ds_load_b128, bank-conflict-free
            v8bf bh  = *(const v8bf*)(bpix + 2*c + 32);
            v8bf a0l = *(const v8bf*)(a0 + c);
            v8bf a0h = *(const v8bf*)(a0 + c + 16);
            v8bf a1l = *(const v8bf*)(a1 + c);
            v8bf a1h = *(const v8bf*)(a1 + c + 16);
            v16bf bf  = __builtin_shufflevector(bl,  bh,  0,1,2,3,4,5,6,7,8,9,10,11,12,13,14,15);
            v16bf af0 = __builtin_shufflevector(a0l, a0h, 0,1,2,3,4,5,6,7,8,9,10,11,12,13,14,15);
            v16bf af1 = __builtin_shufflevector(a1l, a1h, 0,1,2,3,4,5,6,7,8,9,10,11,12,13,14,15);
            acc0 = __builtin_amdgcn_wmma_f32_16x16x32_bf16(false, af0, false, bf,
                                                           (short)0, acc0, false, false);
            acc1 = __builtin_amdgcn_wmma_f32_16x16x32_bf16(false, af1, false, bf,
                                                           (short)0, acc1, false, false);
        }
    }

    // D layout: VGPR j -> row j + 8*hi, col = lane%16. Coalesced 64B f32 stores + bias.
    for (int j = 0; j < 8; ++j) {
        const int m  = m0 + j + hi*8;
        const int m2 = m0 + 16 + j + hi*8;
        out[(size_t)(b*COUT + m )*PIX + p] = acc0[j] + bias[m ];
        out[(size_t)(b*COUT + m2)*PIX + p] = acc1[j] + bias[m2];
    }
}

extern "C" void kernel_launch(void* const* d_in, const int* in_sizes, int n_in,
                              void* d_out, int out_size, void* d_ws, size_t ws_size,
                              hipStream_t stream) {
    (void)in_sizes; (void)n_in; (void)out_size; (void)ws_size;
    const float* inputs  = (const float*)d_in[0];
    const float* route_w = (const float*)d_in[1];
    const float* route_b = (const float*)d_in[2];
    const float* weight  = (const float*)d_in[3];
    const float* bias    = (const float*)d_in[4];
    float* out = (float*)d_out;

    // workspace: gates (1KB) | padded NHWC bf16 (14.7MB) | dyn weights bf16 (37.7MB)
    float*  gates = (float*)d_ws;
    __bf16* xp    = (__bf16*)((char*)d_ws + 1024);
    __bf16* wdyn  = (__bf16*)((char*)d_ws + 1024 + (size_t)BB*XPS*sizeof(__bf16));

    gates_kernel<<<BB, CIN, 0, stream>>>(inputs, route_w, route_b, gates);
    nhwc_kernel <<<BB*30, CIN, 0, stream>>>(inputs, xp);
    wsyn_kernel <<<WPS/256, 256, 0, stream>>>(weight, gates, wdyn);   // after gates (same stream)
    cconv_wmma_kernel<<<BB*49, 256, 0, stream>>>(xp, wdyn, bias, out);
}
